// RealformerAttention_58995670778083
// MI455X (gfx1250) — compile-verified, matrix operand order checked by reading, not run on hardware
//
#include <hip/hip_runtime.h>
#include <hip/hip_bf16.h>

typedef __attribute__((ext_vector_type(16))) __bf16 v16bf;
typedef __attribute__((ext_vector_type(8)))  float  v8f;

#define HIDDEN 1024
#define NH 16
#define DH 64
#define BATCH 2
#define SEQ 2048
#define ROWS (BATCH * SEQ) /* 4096 */

#if defined(__gfx1250__) && __has_builtin(__builtin_amdgcn_global_load_async_to_lds_b128) && \
    __has_builtin(__builtin_amdgcn_s_wait_asynccnt)
#define HAS_ASYNC 1
#else
#define HAS_ASYNC 0
#endif

union Frag {
    v16bf v;
    unsigned u[8];
    __bf16 h[16];
};

// ---- bf16 helpers (bit-level RNE conversion) ----
__device__ __forceinline__ __bf16 f2bf(float f) {
    union { float f; unsigned u; } x; x.f = f;
    unsigned r = (x.u + 0x7FFFu + ((x.u >> 16) & 1u)) >> 16;
    union { unsigned short s; __bf16 h; } y; y.s = (unsigned short)r;
    return y.h;
}
__device__ __forceinline__ unsigned pack2(float a, float b) {
    union { __bf16 h[2]; unsigned u; } t;
    t.h[0] = f2bf(a); t.h[1] = f2bf(b);
    return t.u;
}

// WMMA 16x16x32 fragment layout (CDNA5 ISA 7.12.2):
// lane&15 -> M(A)/N(B); lane>=16 shifts K by 8; element i -> k = (i&7) + 16*(i>>3).
// => elements 0..7 are 8 contiguous k at kbase, elements 8..15 at kbase+16: two b128 loads.
__device__ __forceinline__ v16bf load_frag_g(const __bf16* base, int row, int ld, int k0, int lane) {
    const unsigned* p = (const unsigned*)(base + (size_t)row * ld + k0 + ((lane >> 4) << 3));
    uint4 lo = *(const uint4*)p;
    uint4 hi = *(const uint4*)(p + 8);
    Frag f;
    f.u[0] = lo.x; f.u[1] = lo.y; f.u[2] = lo.z; f.u[3] = lo.w;
    f.u[4] = hi.x; f.u[5] = hi.y; f.u[6] = hi.z; f.u[7] = hi.w;
    return f.v;
}

__device__ __forceinline__ float wred_max(float v) {
#pragma unroll
    for (int o = 16; o > 0; o >>= 1) v = fmaxf(v, __shfl_xor(v, o, 32));
    return v;
}
__device__ __forceinline__ float wred_sum(float v) {
#pragma unroll
    for (int o = 16; o > 0; o >>= 1) v += __shfl_xor(v, o, 32);
    return v;
}

#if HAS_ASYNC
// ROCm 7.2 signature (from diagnostic): param0 = v4i in device-global AS, param1 = LDS side.
typedef int vi4 __attribute__((vector_size(16)));
typedef __attribute__((address_space(1))) vi4 gvi4;
typedef __attribute__((address_space(3))) vi4 lvi4;

template <int N> __device__ __forceinline__ void wait_async() {
    __builtin_amdgcn_s_wait_asynccnt(N);
}
__device__ __forceinline__ void async_ld16(const float* src, float4* dst) {
    __builtin_amdgcn_global_load_async_to_lds_b128(
        (gvi4*)(void*)const_cast<float*>(src),
        (lvi4*)(void*)dst, 0, 0);
}
#endif

// ---------------- fp32 -> bf16 conversion ----------------
__global__ void cvt_bf16(const float* __restrict__ s, __bf16* __restrict__ d, int n) {
    for (int i = blockIdx.x * blockDim.x + threadIdx.x; i < n; i += gridDim.x * blockDim.x)
        d[i] = f2bf(s[i]);
}

// ---------------- fused QKV projection: C = A @ W^T + b ----------------
// A: [4096,1024] bf16. W: [1024,1024] bf16 ([n,k], used transposed). blockIdx.z: 0=Q,1=K,2=V.
// Q,K written [b,h,s,d]; V written TRANSPOSED [b,h,d,s] so attention B-fragments are contiguous.
__global__ void qkv_gemm(const __bf16* __restrict__ A,
                         const __bf16* __restrict__ Wq, const __bf16* __restrict__ Wk,
                         const __bf16* __restrict__ Wv,
                         const float* __restrict__ bq, const float* __restrict__ bk,
                         const float* __restrict__ bv,
                         __bf16* __restrict__ Qo, __bf16* __restrict__ Ko,
                         __bf16* __restrict__ Vt) {
    int which = blockIdx.z;
    const __bf16* W   = (which == 0) ? Wq : (which == 1) ? Wk : Wv;
    const float*  bia = (which == 0) ? bq : (which == 1) ? bk : bv;
    __bf16*       Out = (which == 0) ? Qo : (which == 1) ? Ko : Vt;

    int lane = threadIdx.x & 31;
    int wave = threadIdx.x >> 5;
    int tile = blockIdx.x * 8 + wave;          // 16384 tiles
    int tm = tile / (HIDDEN / 16);
    int tn = tile % (HIDDEN / 16);

    int arow = tm * 16 + (lane & 15);
    int brow = tn * 16 + (lane & 15);

    v8f acc = {};
    for (int k0 = 0; k0 < HIDDEN; k0 += 32) {
        __builtin_prefetch((const void*)(A + (size_t)arow * HIDDEN + k0 + 64), 0, 1);
        v16bf a = load_frag_g(A, arow, HIDDEN, k0, lane);
        v16bf b = load_frag_g(W, brow, HIDDEN, k0, lane);
        acc = __builtin_amdgcn_wmma_f32_16x16x32_bf16(false, a, false, b, (short)0, acc,
                                                      false, false);
    }

    int n = tn * 16 + (lane & 15);
    int h = n / DH, d = n % DH;
    float bb = bia[n];
#pragma unroll
    for (int j = 0; j < 8; ++j) {
        int m = tm * 16 + j + ((lane >> 4) << 3);
        int bi = m / SEQ, s = m % SEQ;
        size_t bh = (size_t)(bi * NH + h);
        __bf16 val = f2bf(acc[j] + bb);
        if (which == 2)
            Out[(bh * DH + d) * SEQ + s] = val;     // V transposed [b,h,d,s]
        else
            Out[(bh * SEQ + s) * DH + d] = val;     // Q,K [b,h,s,d]
    }
}

// ---------------- attention core: one block per (b*nh, 16-row q-tile) ----------------
// LDS: 16 x 2048 fp32 scores (128 KB) + 12 KB async staging.
__global__ void attention(const __bf16* __restrict__ Q, const __bf16* __restrict__ K,
                          const __bf16* __restrict__ Vt, const float* __restrict__ prev,
                          __bf16* __restrict__ Ctx) {
    extern __shared__ float sc[];              // [16][SEQ] then staging
    int lane = threadIdx.x & 31;
    int wave = threadIdx.x >> 5;
    int qt = blockIdx.x % (SEQ / 16);
    int bh = blockIdx.x / (SEQ / 16);          // b*NH + h
    const __bf16* Qb  = Q  + (size_t)bh * SEQ * DH;
    const __bf16* Kb  = K  + (size_t)bh * SEQ * DH;
    const __bf16* Vtb = Vt + (size_t)bh * DH * SEQ;

    // ---- Phase 1: scores = Q Kt / 8, 8 waves stripe across 128 key tiles ----
    int qrow = qt * 16 + (lane & 15);
    v16bf a0 = load_frag_g(Qb, qrow, DH, 0, lane);
    v16bf a1 = load_frag_g(Qb, qrow, DH, 32, lane);
    for (int kt = wave; kt < SEQ / 16; kt += 8) {
        int krow = kt * 16 + (lane & 15);
        v16bf b0 = load_frag_g(Kb, krow, DH, 0, lane);
        v16bf b1 = load_frag_g(Kb, krow, DH, 32, lane);
        v8f c = {};
        c = __builtin_amdgcn_wmma_f32_16x16x32_bf16(false, a0, false, b0, (short)0, c, false, false);
        c = __builtin_amdgcn_wmma_f32_16x16x32_bf16(false, a1, false, b1, (short)0, c, false, false);
#pragma unroll
        for (int j = 0; j < 8; ++j) {
            int r = j + ((lane >> 4) << 3);
            sc[r * SEQ + kt * 16 + (lane & 15)] = c[j] * 0.125f;
        }
    }
    __syncthreads();

    // ---- Phase 2: softmax -> +prev (single 536MB stream, float4/async) -> softmax ----
#if HAS_ASYNC
    float4* stage = (float4*)(sc + 16 * SEQ);  // 8 waves * 3 slots * 32 lanes
#endif
#pragma unroll
    for (int rr = 0; rr < 2; ++rr) {
        int r = wave + rr * 8;
        float*  row  = sc + r * SEQ;
        float4* rowv = (float4*)row;
        const float* prow = prev + (((size_t)bh) * SEQ + (qt * 16 + r)) * SEQ;

        float m1 = -1e30f;
        for (int i = 0; i < 16; ++i) {
            float4 x = rowv[lane + 32 * i];
            m1 = fmaxf(fmaxf(m1, fmaxf(x.x, x.y)), fmaxf(x.z, x.w));
        }
        m1 = wred_max(m1);

        float s1 = 0.f;
        for (int i = 0; i < 16; ++i) {
            int j = lane + 32 * i;
            float4 x = rowv[j], e;
            e.x = __expf(x.x - m1); e.y = __expf(x.y - m1);
            e.z = __expf(x.z - m1); e.w = __expf(x.w - m1);
            s1 += (e.x + e.y) + (e.z + e.w);
            rowv[j] = e;                        // keep unnormalized exp
        }
        s1 = wred_sum(s1);
        float inv1 = 1.f / s1;

        float m2 = -1e30f;
#if HAS_ASYNC
        {
            float4* buf = stage + wave * 96;    // 3 slots x 32 lanes
            async_ld16(prow + 0 * 128 + lane * 4, buf + 0 * 32 + lane);
            async_ld16(prow + 1 * 128 + lane * 4, buf + 1 * 32 + lane);
            for (int i = 0; i < 16; ++i) {
                if (i == 15) wait_async<0>(); else wait_async<1>();
                float4 pv = buf[(i % 3) * 32 + lane];
                int j = lane + 32 * i;
                float4 e = rowv[j], t;
                t.x = e.x * inv1 + pv.x; t.y = e.y * inv1 + pv.y;
                t.z = e.z * inv1 + pv.z; t.w = e.w * inv1 + pv.w;
                rowv[j] = t;
                m2 = fmaxf(fmaxf(m2, fmaxf(t.x, t.y)), fmaxf(t.z, t.w));
                if (i + 2 < 16)
                    async_ld16(prow + (i + 2) * 128 + lane * 4, buf + ((i + 2) % 3) * 32 + lane);
            }
        }
#else
        {
            const float4* prowv = (const float4*)prow;
            for (int i = 0; i < 16; ++i) {
                int j = lane + 32 * i;
                float4 e = rowv[j], pv = prowv[j], t;
                t.x = e.x * inv1 + pv.x; t.y = e.y * inv1 + pv.y;
                t.z = e.z * inv1 + pv.z; t.w = e.w * inv1 + pv.w;
                rowv[j] = t;
                m2 = fmaxf(fmaxf(m2, fmaxf(t.x, t.y)), fmaxf(t.z, t.w));
            }
        }
#endif
        m2 = wred_max(m2);

        float s2 = 0.f;
        for (int i = 0; i < 16; ++i) {
            float4 t = rowv[lane + 32 * i];
            s2 += (__expf(t.x - m2) + __expf(t.y - m2)) +
                  (__expf(t.z - m2) + __expf(t.w - m2));
        }
        s2 = wred_sum(s2);
        float inv2 = 1.f / s2;

        // final probs packed bf16, written IN PLACE into the low half of the row.
        // Lockstep-safe: iter i destroys f32 cols <= 64i+63, iter i+1 reads cols >= 128i+128.
        uint2* rowp = (uint2*)row;
        for (int i = 0; i < 16; ++i) {
            int j = lane + 32 * i;
            float4 t = rowv[j], p;
            p.x = __expf(t.x - m2) * inv2; p.y = __expf(t.y - m2) * inv2;
            p.z = __expf(t.z - m2) * inv2; p.w = __expf(t.w - m2) * inv2;
            uint2 w; w.x = pack2(p.x, p.y); w.y = pack2(p.z, p.w);
            rowp[j] = w;
        }
    }
    __syncthreads();

    // ---- Phase 3: ctx = probs @ V ; waves 0..3 own one 16x16 (q x d) tile each ----
    if (wave < 4) {
        const unsigned* pu = (const unsigned*)sc;   // packed bf16 probs, row stride 2048 words
        v8f acc = {};
        int m = lane & 15;
        int dcol = wave * 16 + (lane & 15);
        for (int k0 = 0; k0 < SEQ; k0 += 32) {
            int w = m * SEQ + (k0 >> 1) + ((lane >> 4) << 2);
            uint4 lo = *(const uint4*)(pu + w);
            uint4 hi = *(const uint4*)(pu + w + 8);
            Frag fa;
            fa.u[0] = lo.x; fa.u[1] = lo.y; fa.u[2] = lo.z; fa.u[3] = lo.w;
            fa.u[4] = hi.x; fa.u[5] = hi.y; fa.u[6] = hi.z; fa.u[7] = hi.w;
            v16bf fb = load_frag_g(Vtb, dcol, SEQ, k0, lane);   // contiguous: Vt is [d,s]
            acc = __builtin_amdgcn_wmma_f32_16x16x32_bf16(false, fa.v, false, fb, (short)0,
                                                          acc, false, false);
        }
        int h = bh % NH, b = bh / NH;
#pragma unroll
        for (int j = 0; j < 8; ++j) {
            int qpos = qt * 16 + j + ((lane >> 4) << 3);
            Ctx[((size_t)(b * SEQ) + qpos) * HIDDEN + h * DH + dcol] = f2bf(acc[j]);
        }
    }
}

// ---------------- output projection: out = ctx @ Wo^T + bo (fp32 out) ----------------
__global__ void out_gemm(const __bf16* __restrict__ A, const __bf16* __restrict__ W,
                         const float* __restrict__ bias, float* __restrict__ Out) {
    int lane = threadIdx.x & 31;
    int wave = threadIdx.x >> 5;
    int tile = blockIdx.x * 8 + wave;
    int tm = tile / (HIDDEN / 16);
    int tn = tile % (HIDDEN / 16);

    int arow = tm * 16 + (lane & 15);
    int brow = tn * 16 + (lane & 15);
    v8f acc = {};
    for (int k0 = 0; k0 < HIDDEN; k0 += 32) {
        __builtin_prefetch((const void*)(A + (size_t)arow * HIDDEN + k0 + 64), 0, 1);
        v16bf a = load_frag_g(A, arow, HIDDEN, k0, lane);
        v16bf b = load_frag_g(W, brow, HIDDEN, k0, lane);
        acc = __builtin_amdgcn_wmma_f32_16x16x32_bf16(false, a, false, b, (short)0, acc,
                                                      false, false);
    }
    int n = tn * 16 + (lane & 15);
    float bb = bias[n];
#pragma unroll
    for (int j = 0; j < 8; ++j) {
        int m = tm * 16 + j + ((lane >> 4) << 3);
        Out[(size_t)m * HIDDEN + n] = acc[j] + bb;
    }
}

extern "C" void kernel_launch(void* const* d_in, const int* in_sizes, int n_in,
                              void* d_out, int out_size, void* d_ws, size_t ws_size,
                              hipStream_t stream) {
    const float* hidden = (const float*)d_in[0];
    const float* prev   = (const float*)d_in[1];
    const float* Wq = (const float*)d_in[2];
    const float* bq = (const float*)d_in[3];
    const float* Wk = (const float*)d_in[4];
    const float* bk = (const float*)d_in[5];
    const float* Wv = (const float*)d_in[6];
    const float* bv = (const float*)d_in[7];
    const float* Wo = (const float*)d_in[8];
    const float* bo = (const float*)d_in[9];
    float* out = (float*)d_out;

    const size_t HH = (size_t)HIDDEN * HIDDEN;
    const size_t RH = (size_t)ROWS * HIDDEN;
    char* p = (char*)d_ws;
    __bf16* hs  = (__bf16*)p; p += RH * 2;
    __bf16* wq  = (__bf16*)p; p += HH * 2;
    __bf16* wk  = (__bf16*)p; p += HH * 2;
    __bf16* wv  = (__bf16*)p; p += HH * 2;
    __bf16* wo  = (__bf16*)p; p += HH * 2;
    __bf16* q   = (__bf16*)p; p += RH * 2;
    __bf16* k   = (__bf16*)p; p += RH * 2;
    __bf16* vt  = (__bf16*)p; p += RH * 2;           // V transposed [b,h,d,s]
    __bf16* ctx = (__bf16*)p; p += RH * 2;           // ~48 MB total

    cvt_bf16<<<1024, 256, 0, stream>>>(hidden, hs, (int)RH);
    cvt_bf16<<<512, 256, 0, stream>>>(Wq, wq, (int)HH);
    cvt_bf16<<<512, 256, 0, stream>>>(Wk, wk, (int)HH);
    cvt_bf16<<<512, 256, 0, stream>>>(Wv, wv, (int)HH);
    cvt_bf16<<<512, 256, 0, stream>>>(Wo, wo, (int)HH);

    qkv_gemm<<<dim3(2048, 1, 3), 256, 0, stream>>>(hs, wq, wk, wv, bq, bk, bv, q, k, vt);

    size_t lds = (size_t)16 * SEQ * sizeof(float) + 8 * 3 * 32 * sizeof(float4);
    attention<<<dim3(BATCH * NH * (SEQ / 16)), 256, lds, stream>>>(q, k, vt, prev, ctx);

    out_gemm<<<2048, 256, 0, stream>>>(ctx, wo, bo, out);
}
